// LlamaAttention_18124761989598
// MI455X (gfx1250) — matrix operand order repaired and checked by reading, not compiled, not run on hardware
//
#include <hip/hip_runtime.h>

// ---------------------------------------------------------------------------
// Types for CDNA5 WMMA (wave32, 16x16x32 bf16 -> f32)
// ---------------------------------------------------------------------------
typedef __bf16 bf16_t;
typedef bf16_t v16bf __attribute__((ext_vector_type(16)));
typedef float  v8f   __attribute__((ext_vector_type(8)));
typedef unsigned int u32x4 __attribute__((ext_vector_type(4)));
typedef int          i32x8 __attribute__((ext_vector_type(8)));
typedef int          i32x4 __attribute__((ext_vector_type(4)));

#if __has_builtin(__builtin_amdgcn_tensor_load_to_lds) && \
    __has_builtin(__builtin_amdgcn_s_wait_tensorcnt)
#define HAVE_TDM 1
#else
#define HAVE_TDM 0
#endif

union FragB16 {
  v16bf v;
  unsigned u[8];
  unsigned short h[16];
};

__device__ __forceinline__ bf16_t f2bf(float f) {
  union { float f; unsigned u; } a; a.f = f;
  unsigned r = (a.u + 0x7FFFu + ((a.u >> 16) & 1u)) >> 16;
  unsigned short hs = (unsigned short)r;
  bf16_t b;
  __builtin_memcpy(&b, &hs, sizeof(b));
  return b;
}

#if HAVE_TDM
// ---------------------------------------------------------------------------
// 2D TDM load: tile_d1 rows x tile_d0 bf16 (contiguous) from global
// (row stride = stride_elems) into LDS at lds_byte_off (packed rows).
// D# packing per CDNA5 ISA.
// ---------------------------------------------------------------------------
__device__ __forceinline__ void tdm_load_2d(unsigned lds_byte_off, const void* gaddr,
                                            unsigned tile_d0, unsigned tile_d1,
                                            unsigned long long stride_elems,
                                            unsigned tensor_d0, unsigned tensor_d1) {
  unsigned long long ga = (unsigned long long)gaddr;
  u32x4 g0;
  g0.x = 1u;                                                // count=1, user-mode D#
  g0.y = lds_byte_off;                                      // lds_addr (bytes)
  g0.z = (unsigned)(ga & 0xFFFFFFFFu);                      // global_addr[31:0]
  g0.w = (unsigned)((ga >> 32) & 0x01FFFFFFu) | (2u << 30); // global_addr[56:32]|type=2
  i32x8 g1;
  g1[0] = (int)(1u << 16);                                  // data_size=1 (2 bytes)
  g1[1] = (int)((tensor_d0 & 0xFFFFu) << 16);               // tensor_dim0[15:0]
  g1[2] = (int)(((tensor_d0 >> 16) & 0xFFFFu) | ((tensor_d1 & 0xFFFFu) << 16));
  g1[3] = (int)(((tensor_d1 >> 16) & 0xFFFFu) | ((tile_d0 & 0xFFFFu) << 16));
  g1[4] = (int)(tile_d1 & 0xFFFFu);                         // tile_dim1 | tile_dim2=0
  g1[5] = (int)(stride_elems & 0xFFFFFFFFull);              // dim0_stride[31:0]
  g1[6] = (int)((stride_elems >> 32) & 0xFFFFull);          // dim0_stride[47:32]
  g1[7] = 0;
  i32x4 z4 = {0, 0, 0, 0};
#if defined(__clang_major__) && __clang_major__ >= 23
  i32x8 z8 = {0, 0, 0, 0, 0, 0, 0, 0};
  __builtin_amdgcn_tensor_load_to_lds(g0, g1, z4, z4, z8, 0);
#else
  __builtin_amdgcn_tensor_load_to_lds(g0, g1, z4, z4, 0);
#endif
}
#endif  // HAVE_TDM

// ---------------------------------------------------------------------------
// Elementwise fp32 -> bf16 cast
// ---------------------------------------------------------------------------
__global__ void cast_kernel(const float* __restrict__ in, bf16_t* __restrict__ out, int n) {
  int i = blockIdx.x * blockDim.x + threadIdx.x;
  if (i < n) out[i] = f2bf(in[i]);
}

// ---------------------------------------------------------------------------
// fp32 [R,C] -> bf16 [C,R] transpose (LDS tiled 32x32, padded)
// ---------------------------------------------------------------------------
__global__ void castT_kernel(const float* __restrict__ in, bf16_t* __restrict__ out,
                             int R, int C) {
  __shared__ float tile[32][33];
  int c0 = blockIdx.x * 32;
  int r0 = blockIdx.y * 32;
  int tx = threadIdx.x;
  int ty = threadIdx.y;
  #pragma unroll
  for (int i = ty; i < 32; i += 8)
    tile[i][tx] = in[(size_t)(r0 + i) * C + (c0 + tx)];
  __syncthreads();
  #pragma unroll
  for (int i = ty; i < 32; i += 8)
    out[(size_t)(c0 + i) * R + (r0 + tx)] = f2bf(tile[tx][i]);
}

// ---------------------------------------------------------------------------
// bf16 transpose of the V slab: in = QKVb V-section, [2048 keys x 1024 dims]
// with row stride 6144 -> Vt [1024 dims x 2048 keys], row stride 2048.
// Makes the PV B-fragments contiguous (b128 loads instead of u16 gathers).
// ---------------------------------------------------------------------------
__global__ void transposeV_kernel(const bf16_t* __restrict__ in, bf16_t* __restrict__ out) {
  __shared__ bf16_t t[32][33];
  int c0 = blockIdx.x * 32;   // dim index
  int r0 = blockIdx.y * 32;   // key index
  int tx = threadIdx.x;
  int ty = threadIdx.y;
  #pragma unroll
  for (int i = ty; i < 32; i += 8)
    t[i][tx] = in[(size_t)(r0 + i) * 6144 + (c0 + tx)];
  __syncthreads();
  #pragma unroll
  for (int i = ty; i < 32; i += 8)
    out[(size_t)(c0 + i) * 2048 + (r0 + tx)] = t[tx][i];
}

// ---------------------------------------------------------------------------
// WMMA GEMM: C[M,N] = A[M,K] * Bt[N,K]^T.
// 256 threads = 8 waves; block tile = 256(M) x 64(N); wave tile = 32 x 64
// (two 16-row A tiles reuse each B fragment -> 8 WMMA / wave / k-step).
// B tile (64 x 32 bf16) staged in LDS by the Tensor Data Mover,
// double-buffered; sync = s_wait_tensorcnt 0 + workgroup barrier.
// ---------------------------------------------------------------------------
template <bool F32OUT>
__global__ __launch_bounds__(256, 1) void gemm_kernel(
    const bf16_t* __restrict__ A, const bf16_t* __restrict__ Bt,
    void* __restrict__ C, int M, int N, int K) {
  __shared__ __align__(16) bf16_t Bld[2][64][32];   // 2 x 4 KB double buffer

  int wave = threadIdx.x >> 5;
  int lane = threadIdx.x & 31;
  int half = lane >> 4;
  int col  = lane & 15;
  int m0 = blockIdx.y * 256 + wave * 32;
  int n0 = blockIdx.x * 64;

  const bf16_t* arow0 = A + (size_t)(m0 + col) * K;
  const bf16_t* arow1 = A + (size_t)(m0 + 16 + col) * K;
  v8f acc[2][4] = {};

  // ---- prologue: stage first B tile into buffer 0 ----
#if HAVE_TDM
  if (wave == 0)
    tdm_load_2d((unsigned)(unsigned long long)(void*)&Bld[0][0][0],
                Bt + (size_t)n0 * K, 32u, 64u,
                (unsigned long long)K, (unsigned)K, (unsigned)N);
#else
  {
    int r_ = threadIdx.x >> 2, s_ = (threadIdx.x & 3) * 8;
    *(int4*)&Bld[0][r_][s_] = *(const int4*)(Bt + (size_t)(n0 + r_) * K + s_);
  }
#endif

  int buf = 0;
  for (int k0 = 0; k0 < K; k0 += 32) {
#if HAVE_TDM
    if (wave == 0) __builtin_amdgcn_s_wait_tensorcnt(0);
#endif
    __syncthreads();
    if (k0 + 32 < K) {
#if HAVE_TDM
      if (wave == 0)
        tdm_load_2d((unsigned)(unsigned long long)(void*)&Bld[buf ^ 1][0][0],
                    Bt + (size_t)n0 * K + (k0 + 32), 32u, 64u,
                    (unsigned long long)K, (unsigned)K, (unsigned)N);
#else
      int r_ = threadIdx.x >> 2, s_ = (threadIdx.x & 3) * 8;
      *(int4*)&Bld[buf ^ 1][r_][s_] =
          *(const int4*)(Bt + (size_t)(n0 + r_) * K + (k0 + 32) + s_);
#endif
    }

    FragB16 af[2];
    #pragma unroll
    for (int v = 0; v < 8; ++v) {
      int kb = ((v < 4) ? 0 : 16) + half * 8 + (v & 3) * 2;
      af[0].u[v] = *(const unsigned*)(arow0 + k0 + kb);
      af[1].u[v] = *(const unsigned*)(arow1 + k0 + kb);
    }
    __builtin_prefetch(arow0 + k0 + 64, 0, 1);   // global_prefetch_b8
    __builtin_prefetch(arow1 + k0 + 64, 0, 1);

    #pragma unroll
    for (int t = 0; t < 4; ++t) {
      FragB16 bfrag;
      #pragma unroll
      for (int v = 0; v < 8; ++v)
        bfrag.u[v] = *(const unsigned*)&Bld[buf][t * 16 + col][((v < 4) ? 0 : 16) + half * 8 + (v & 3) * 2];
      acc[0][t] = __builtin_amdgcn_wmma_f32_16x16x32_bf16(
          false, af[0].v, false, bfrag.v, (short)0, acc[0][t], false, false);
      acc[1][t] = __builtin_amdgcn_wmma_f32_16x16x32_bf16(
          false, af[1].v, false, bfrag.v, (short)0, acc[1][t], false, false);
    }
    __syncthreads();
    buf ^= 1;
  }

  #pragma unroll
  for (int mt = 0; mt < 2; ++mt) {
    #pragma unroll
    for (int t = 0; t < 4; ++t) {
      int n = n0 + t * 16 + col;
      #pragma unroll
      for (int r = 0; r < 8; ++r) {
        size_t idx = (size_t)(m0 + mt * 16 + r + half * 8) * N + n;
        float val = acc[mt][t][r];
        if (F32OUT) ((float*)C)[idx] = val;
        else        ((bf16_t*)C)[idx] = f2bf(val);
      }
    }
  }
}

// ---------------------------------------------------------------------------
// Flash-style causal GQA attention (wave32).
// qkv: bf16 [2048, 6144] = [Q(32x128) | K(8x128) | V(8x128)]
// Vt:  bf16 [1024, 2048]  (V transposed: dim-major, keys contiguous)
// aout: bf16 [2048, 4096]. Grid (S/64, heads), block 128 = 4 waves.
// ---------------------------------------------------------------------------
__global__ __launch_bounds__(128, 1) void attn_kernel(
    const bf16_t* __restrict__ qkv, const bf16_t* __restrict__ Vt,
    bf16_t* __restrict__ aout) {
  constexpr int QKVN = 6144, D = 4096, HD = 128, S = 2048;
  __shared__ __align__(4) bf16_t Pld[4][16][32];

  int wave = threadIdx.x >> 5;
  int lane = threadIdx.x & 31;
  int half = lane >> 4;
  int col  = lane & 15;
  int head = blockIdx.y;
  int kvh  = head >> 2;
  int q0   = blockIdx.x * 64 + wave * 16;
  const float sm_scale = 0.08838834764831845f;   // 1/sqrt(128)

  FragB16 qf[4];
  const bf16_t* qrow = qkv + (size_t)(q0 + col) * QKVN + head * HD;
  #pragma unroll
  for (int ks = 0; ks < 4; ++ks) {
    #pragma unroll
    for (int v = 0; v < 8; ++v) {
      int kb = ((v < 4) ? 0 : 16) + half * 8 + (v & 3) * 2;
      qf[ks].u[v] = *(const unsigned*)(qrow + ks * 32 + kb);
    }
  }

  v8f o[8] = {};
  float rmax[8], rsum[8];
  #pragma unroll
  for (int r = 0; r < 8; ++r) { rmax[r] = -1e30f; rsum[r] = 0.f; }

  int nk = q0 + 16;
  for (int kb0 = 0; kb0 < nk; kb0 += 32) {
    v8f st[2];
    #pragma unroll
    for (int sub = 0; sub < 2; ++sub) {
      v8f s = {};
      int kt = kb0 + sub * 16;
      const bf16_t* kp = qkv + (size_t)(kt + col) * QKVN + 4096 + kvh * HD;
      #pragma unroll
      for (int ks = 0; ks < 4; ++ks) {
        FragB16 kf;
        #pragma unroll
        for (int v = 0; v < 8; ++v)
          kf.u[v] = *(const unsigned*)(kp + ks * 32 + ((v < 4) ? 0 : 16) + half * 8 + (v & 3) * 2);
        s = __builtin_amdgcn_wmma_f32_16x16x32_bf16(
            false, qf[ks].v, false, kf.v, (short)0, s, false, false);
      }
      int kg = kt + col;
      #pragma unroll
      for (int r = 0; r < 8; ++r) {
        float x = s[r] * sm_scale;
        int qg = q0 + r + half * 8;
        st[sub][r] = (kg <= qg) ? x : -1e30f;
      }
    }

    float scl[8];
    #pragma unroll
    for (int r = 0; r < 8; ++r) {
      float v = fmaxf(st[0][r], st[1][r]);
      v = fmaxf(v, __shfl_xor(v, 1, 32));
      v = fmaxf(v, __shfl_xor(v, 2, 32));
      v = fmaxf(v, __shfl_xor(v, 4, 32));
      v = fmaxf(v, __shfl_xor(v, 8, 32));
      float nm = fmaxf(rmax[r], v);
      scl[r] = __expf(rmax[r] - nm);
      rmax[r] = nm;
      float p0 = __expf(st[0][r] - nm);
      float p1 = __expf(st[1][r] - nm);
      st[0][r] = p0; st[1][r] = p1;
      float ps = p0 + p1;
      ps += __shfl_xor(ps, 1, 32);
      ps += __shfl_xor(ps, 2, 32);
      ps += __shfl_xor(ps, 4, 32);
      ps += __shfl_xor(ps, 8, 32);
      rsum[r] = rsum[r] * scl[r] + ps;
    }
    #pragma unroll
    for (int t = 0; t < 8; ++t)
      #pragma unroll
      for (int r = 0; r < 8; ++r)
        o[t][r] *= scl[r];

    // P: C-layout -> LDS row-major -> reload as A fragment
    #pragma unroll
    for (int r = 0; r < 8; ++r) {
      Pld[wave][r + half * 8][col]      = f2bf(st[0][r]);
      Pld[wave][r + half * 8][16 + col] = f2bf(st[1][r]);
    }
    FragB16 pf;
    #pragma unroll
    for (int v = 0; v < 8; ++v) {
      int kb = ((v < 4) ? 0 : 16) + half * 8 + (v & 3) * 2;
      pf.u[v] = *(const unsigned*)&Pld[wave][col][kb];
    }

    // O += P * V : V^T fragments are contiguous in Vt -> b128 loads
    #pragma unroll
    for (int t = 0; t < 8; ++t) {
      FragB16 vf;
      const bf16_t* vrow = Vt + (size_t)(kvh * HD + t * 16 + col) * S + kb0;
      #pragma unroll
      for (int v = 0; v < 8; ++v)
        vf.u[v] = *(const unsigned*)(vrow + ((v < 4) ? 0 : 16) + half * 8 + (v & 3) * 2);
      o[t] = __builtin_amdgcn_wmma_f32_16x16x32_bf16(
          false, pf.v, false, vf.v, (short)0, o[t], false, false);
    }
  }

  #pragma unroll
  for (int t = 0; t < 8; ++t) {
    int d = head * HD + t * 16 + col;
    #pragma unroll
    for (int r = 0; r < 8; ++r) {
      int m = q0 + r + half * 8;
      aout[(size_t)m * D + d] = f2bf(o[t][r] / rsum[r]);
    }
  }
}

// ---------------------------------------------------------------------------
// Host launcher
// ---------------------------------------------------------------------------
extern "C" void kernel_launch(void* const* d_in, const int* in_sizes, int n_in,
                              void* d_out, int out_size, void* d_ws, size_t ws_size,
                              hipStream_t stream) {
  (void)in_sizes; (void)n_in; (void)out_size; (void)ws_size;
  constexpr int S = 2048, HID = 4096, QKVN = 6144;

  const float* X    = (const float*)d_in[1];
  const float* Wqkv = (const float*)d_in[2];
  const float* Wo   = (const float*)d_in[3];
  float* out = (float*)d_out;

  char* ws = (char*)d_ws;
  size_t off = 0;
  bf16_t* Xb    = (bf16_t*)(ws + off); off += (size_t)S * HID * 2;      // 16 MB
  bf16_t* WqkvT = (bf16_t*)(ws + off); off += (size_t)QKVN * HID * 2;   // 48 MB
  bf16_t* WoT   = (bf16_t*)(ws + off); off += (size_t)HID * HID * 2;    // 32 MB
  bf16_t* QKVb  = (bf16_t*)(ws + off); off += (size_t)S * QKVN * 2;     // 24 MB
  bf16_t* Vt    = (bf16_t*)(ws + off); off += (size_t)1024 * S * 2;     //  4 MB
  bf16_t* AOb   = Xb;   // Xb dead after QKV GEMM; reuse

  cast_kernel<<<(S * HID) / 256, 256, 0, stream>>>(X, Xb, S * HID);
  castT_kernel<<<dim3(QKVN / 32, HID / 32), dim3(32, 8), 0, stream>>>(Wqkv, WqkvT, HID, QKVN);
  castT_kernel<<<dim3(HID / 32, HID / 32), dim3(32, 8), 0, stream>>>(Wo, WoT, HID, HID);

  gemm_kernel<false><<<dim3(QKVN / 64, S / 256), 256, 0, stream>>>(
      Xb, WqkvT, (void*)QKVb, S, QKVN, HID);

  transposeV_kernel<<<dim3(1024 / 32, S / 32), dim3(32, 8), 0, stream>>>(
      QKVb + 5120, Vt);

  attn_kernel<<<dim3(S / 64, 32), 128, 0, stream>>>(QKVb, Vt, AOb);

  gemm_kernel<true><<<dim3(HID / 64, S / 256), 256, 0, stream>>>(
      AOb, WoT, (void*)out, S, HID, HID);
}
